// QConvModel_38843684225835
// MI455X (gfx1250) — matrix operand order repaired and checked by reading, not compiled, not run on hardware
//
#include <hip/hip_runtime.h>
#include <cstdint>

// ---------------------------------------------------------------------------
// QConvModel on MI455X (gfx1250), wave32 + WMMA.
//
// Roofline: fc6's weight matrix (128 x 524288 f32 = 268 MB > 192 MB L2) must
// stream once from HBM -> ~11.5 us at 23.3 TB/s; that bounds the whole net.
// conv1 (~2.4 GFLOP) runs on v_wmma_f32_16x16x32_f16 and is sub-us.
//
//   k_minmax   : p_sf = clip(max|x|,1e-8)/63                        (1 block)
//   k_quant_x  : x_int = clip(rint(x/p_sf)) -> f16 (exact integers)
//   k_prep_w0/2: per-OC weight scales + integer weights -> f16 (exact)
//   k_conv0    : implicit GEMM M=64 K=27->32 N=16384; LDS patch; 1 WMMA/wave;
//                A fragment = two b128 loads (K pattern is 2x8 contiguous)
//   k_split    : xs = h1/p_sf -> (hi,lo) f16 split (~22 mantissa bits)
//   k_conv1    : implicit GEMM M=128 K=576 N=16384; LDS-staged patch;
//                2 WMMAs per 32-wide K tile (hi+lo) = 36 WMMA/wave
//   k_pool     : 2x2 maxpool fused with /p_sf
//   k_fc6      : 1 block/row, 1024 thr: pass1 min/max streams the 2 MB row
//                (HBM), pass2 quantized dot re-reads it L2-resident; b128
//                loads + global_prefetch_b8
//   k_fc8      : tiny 5x128 quantized GEMV -> d_out[5]
// ---------------------------------------------------------------------------

typedef __attribute__((ext_vector_type(16))) _Float16 v16h;
typedef __attribute__((ext_vector_type(8)))  _Float16 v8h;
typedef __attribute__((ext_vector_type(8)))  float    v8f;

#define QMAXF  63.0f   // 2^(7-1)-1
#define QMINF -64.0f

__device__ __forceinline__ float qround_clip(float v) {
  // jnp.round == round-half-to-even == rintf (default RNE)
  return fminf(fmaxf(rintf(v), QMINF), QMAXF);
}

// Two contiguous b128 loads give one lane's A fragment: elements 0..7 are
// K = base..base+7, elements 8..15 are K = base+16..base+23 (ISA 7.12.2).
__device__ __forceinline__ v16h load_a_frag(const _Float16* p) {
  const v8h* pv = (const v8h*)p;
  v8h lo = pv[0];
  v8h hi = pv[2];
  return __builtin_shufflevector(lo, hi, 0, 1, 2, 3, 4, 5, 6, 7,
                                 8, 9, 10, 11, 12, 13, 14, 15);
}

// ------------------------------- scales ------------------------------------

__global__ void k_minmax(const float* __restrict__ x, int n, float* __restrict__ sc) {
  __shared__ float rmn[1024], rmx[1024];
  const int tid = threadIdx.x;
  float mn = __builtin_inff(), mx = -__builtin_inff();
  for (int i = tid; i < n; i += 1024) {
    float v = x[i];
    mn = fminf(mn, v);
    mx = fmaxf(mx, v);
  }
  rmn[tid] = mn; rmx[tid] = mx;
  __syncthreads();
  for (int s = 512; s > 0; s >>= 1) {
    if (tid < s) {
      rmn[tid] = fminf(rmn[tid], rmn[tid + s]);
      rmx[tid] = fmaxf(rmx[tid], rmx[tid + s]);
    }
    __syncthreads();
  }
  if (tid == 0) {
    float a = fmaxf(fabsf(rmn[0]), fabsf(rmx[0]));
    float p = fmaxf(a, 1e-8f) / 63.0f;
    sc[0] = p;
    sc[1] = 1.0f / p;
  }
}

__global__ void k_quant_x(const float* __restrict__ x, const float* __restrict__ sc,
                          _Float16* __restrict__ xq, int n) {
  int i = blockIdx.x * blockDim.x + threadIdx.x;
  if (i < n) {
    float p = sc[0];
    xq[i] = (_Float16)qround_clip(x[i] / p);   // integer in [-64,63]: exact in f16
  }
}

// One wave per output channel: butterfly min/max over 27 taps, quantize, pad K->32.
__global__ void k_prep_w0(const float* __restrict__ w, const float* __restrict__ bsrc,
                          const float* __restrict__ sc,
                          _Float16* __restrict__ wq, float* __restrict__ b_int,
                          float* __restrict__ b_sf) {
  const int c = blockIdx.x, l = threadIdx.x;      // 32 lanes (wave32)
  float v  = (l < 27) ? w[c * 27 + l] : 0.0f;
  float mn = (l < 27) ? v :  __builtin_inff();
  float mx = (l < 27) ? v : -__builtin_inff();
  for (int m = 16; m >= 1; m >>= 1) {
    mn = fminf(mn, __shfl_xor(mn, m, 32));
    mx = fmaxf(mx, __shfl_xor(mx, m, 32));
  }
  float wsf = fmaxf(fmaxf(fabsf(mn), fabsf(mx)), 1e-8f) / 63.0f;
  wq[c * 32 + l] = (_Float16)((l < 27) ? qround_clip(v / wsf) : 0.0f);
  if (l == 0) {
    float bs = wsf * sc[0];
    b_sf[c]  = bs;
    b_int[c] = qround_clip(bsrc[c] / bs);
  }
}

// One wave per output channel: 576 = 32 lanes x 18 taps.
__global__ void k_prep_w2(const float* __restrict__ w, const float* __restrict__ bsrc,
                          const float* __restrict__ sc,
                          _Float16* __restrict__ wq, float* __restrict__ b_int,
                          float* __restrict__ b_sf) {
  const int c = blockIdx.x, l = threadIdx.x;
  float vv[18];
  float mn = __builtin_inff(), mx = -__builtin_inff();
#pragma unroll
  for (int t = 0; t < 18; ++t) {
    float v = w[c * 576 + t * 32 + l];
    vv[t] = v;
    mn = fminf(mn, v);
    mx = fmaxf(mx, v);
  }
  for (int m = 16; m >= 1; m >>= 1) {
    mn = fminf(mn, __shfl_xor(mn, m, 32));
    mx = fmaxf(mx, __shfl_xor(mx, m, 32));
  }
  float wsf = fmaxf(fmaxf(fabsf(mn), fabsf(mx)), 1e-8f) / 63.0f;
#pragma unroll
  for (int t = 0; t < 18; ++t)
    wq[c * 576 + t * 32 + l] = (_Float16)qround_clip(vv[t] / wsf);
  if (l == 0) {
    float bs = wsf * sc[0];
    b_sf[c]  = bs;
    b_int[c] = qround_clip(bsrc[c] / bs);
  }
}

// ------------------------------- conv0 -------------------------------------
// M=64 (oc), K=27 pad 32, N=16384 spatial. 4 waves/block, each wave one
// 16(M)x16(N) tile; one V_WMMA_F32_16X16X32_F16 (exact: both operands are
// small integers). B gathers come from an LDS-staged 3ch x 3 x 18 patch.
__global__ void __launch_bounds__(128)
k_conv0(const _Float16* __restrict__ xq, const _Float16* __restrict__ w0q,
        const float* __restrict__ b_int, const float* __restrict__ b_sf,
        float* __restrict__ h1) {
  __shared__ _Float16 sp[3][3][20];        // [ci][ky][x0-1 .. x0+16], pad->20

  const int tid  = threadIdx.x;            // 128 threads = 4 waves
  const int lane = tid & 31;
  const int wave = tid >> 5;               // 0..3 -> oc tile
  const int m0   = wave * 16;
  const int n0   = blockIdx.x * 16;        // 16 pixels in one image row
  const int half = lane >> 4;
  const int l15  = lane & 15;
  const int y    = n0 >> 7;
  const int x0   = n0 & 127;

  // Cooperative patch fill: 3*3*18 = 162 elements.
  for (int i = tid; i < 3 * 3 * 18; i += 128) {
    int ci = i / 54, t = i % 54, r = t / 18, cx = t % 18;
    int iy = y + r - 1, ix = x0 + cx - 1;
    _Float16 v = (_Float16)0.0f;
    if (iy >= 0 && iy < 128 && ix >= 0 && ix < 128)
      v = xq[ci * 16384 + iy * 128 + ix];
    sp[ci][r][cx] = v;
  }
  __syncthreads();

  // A: two b128 loads (16-bit A 16x32 layout, ISA 7.12.2).
  v16h a = load_a_frag(w0q + (m0 + l15) * 32 + 8 * half);

  // B: 32x16; lane half h -> K 8h..8h+7 then 16+8h..16+8h+7; N = lane&15.
  v16h b;
#pragma unroll
  for (int j = 0; j < 16; ++j) {
    int K = (j & 7) + 8 * half + ((j >> 3) << 4);
    _Float16 v = (_Float16)0.0f;
    if (K < 27) {
      int ci = K / 9, t = K % 9, r = t / 3, s = t % 3;
      v = sp[ci][r][l15 + s];
    }
    b[j] = v;
  }
  v8f acc = {};
  acc = __builtin_amdgcn_wmma_f32_16x16x32_f16(false, a, false, b, (short)0, acc,
                                               false, false);
  // C layout: VGPR r, lanes 0-15 -> M=r, lanes 16-31 -> M=r+8, N=lane&15.
#pragma unroll
  for (int r = 0; r < 8; ++r) {
    int oc = m0 + r + 8 * half;
    h1[oc * 16384 + n0 + l15] = (acc[r] + b_int[oc]) * b_sf[oc];
  }
}

// f16 split of xs = h1 / p_sf: xs ~= hi + lo with ~22 mantissa bits preserved.
__global__ void k_split(const float* __restrict__ h1, const float* __restrict__ sc,
                        _Float16* __restrict__ hi, _Float16* __restrict__ lo, int n) {
  int i = blockIdx.x * blockDim.x + threadIdx.x;
  if (i < n) {
    float xs = h1[i] / sc[0];
    _Float16 h = (_Float16)xs;
    hi[i] = h;
    lo[i] = (_Float16)(xs - (float)h);
  }
}

// ------------------------------- conv1 -------------------------------------
// M=128 (oc), K=576 (64ch x 9), N=16384. 8 waves/block, block = 16 pixels of
// one row. Input patch (64ch x 3 x 18, hi+lo) staged in LDS; 18 K-tiles x
// (WMMA hi + WMMA lo) = 36 v_wmma per wave. A fragments: 2 b128 loads each.
__global__ void __launch_bounds__(256)
k_conv1(const _Float16* __restrict__ hi, const _Float16* __restrict__ lo,
        const _Float16* __restrict__ w2q,
        const float* __restrict__ b_int, const float* __restrict__ b_sf,
        float* __restrict__ h2) {
  __shared__ _Float16 sh[2][64][3][20];    // [hi/lo][ci][ky][x0-1..x0+16], pad->20

  const int tid  = threadIdx.x;            // 256 threads = 8 waves
  const int lane = tid & 31;
  const int wave = tid >> 5;               // 0..7 -> oc tile
  const int m0   = wave * 16;
  const int n0   = blockIdx.x * 16;
  const int half = lane >> 4;
  const int l15  = lane & 15;
  const int y    = n0 >> 7;
  const int x0   = n0 & 127;

  // Cooperative patch load: 64*3*18 = 3456 elements per plane.
  for (int i = tid; i < 64 * 3 * 18; i += 256) {
    int ci = i / 54, t = i % 54, r = t / 18, cx = t % 18;
    int iy = y + r - 1, ix = x0 + cx - 1;
    _Float16 vh = (_Float16)0.0f, vl = (_Float16)0.0f;
    if (iy >= 0 && iy < 128 && ix >= 0 && ix < 128) {
      int g = ci * 16384 + iy * 128 + ix;
      vh = hi[g];
      vl = lo[g];
    }
    sh[0][ci][r][cx] = vh;
    sh[1][ci][r][cx] = vl;
  }
  __syncthreads();

  v8f acc = {};
  const _Float16* wrow = w2q + (m0 + l15) * 576 + 8 * half;
  for (int kt = 0; kt < 18; ++kt) {
    const int k0 = kt * 32;
    v16h a = load_a_frag(wrow + k0);
    v16h bh, bl;
#pragma unroll
    for (int j = 0; j < 16; ++j) {
      int kk = k0 + (j & 7) + 8 * half + ((j >> 3) << 4);
      int ci = kk / 9, t = kk % 9, r = t / 3, s = t % 3;
      bh[j] = sh[0][ci][r][l15 + s];
      bl[j] = sh[1][ci][r][l15 + s];
    }
    acc = __builtin_amdgcn_wmma_f32_16x16x32_f16(false, a, false, bh, (short)0, acc,
                                                 false, false);
    acc = __builtin_amdgcn_wmma_f32_16x16x32_f16(false, a, false, bl, (short)0, acc,
                                                 false, false);
  }

#pragma unroll
  for (int r = 0; r < 8; ++r) {
    int oc = m0 + r + 8 * half;
    h2[oc * 16384 + n0 + l15] = (acc[r] + b_int[oc]) * b_sf[oc];
  }
}

// 2x2 maxpool fused with /p_sf -> xv (fc6 activation "integers").
__global__ void k_pool(const float* __restrict__ h2, const float* __restrict__ sc,
                       float* __restrict__ xv, int n) {
  int o = blockIdx.x * blockDim.x + threadIdx.x;
  if (o < n) {
    int c = o >> 12, rem = o & 4095, py = rem >> 6, px = rem & 63;
    const float* p = h2 + c * 16384 + (py * 2) * 128 + px * 2;
    float m = fmaxf(fmaxf(p[0], p[1]), fmaxf(p[128], p[129]));
    xv[o] = m / sc[0];
  }
}

// ------------------------------- fc6 ---------------------------------------
// One block per output row, 1024 threads (32 waves) to keep many b128 loads
// in flight. Pass 1 streams the 2 MB row from HBM (268 MB across the grid:
// the roofline limiter). Pass 2 re-reads it L2-resident for the quantized dot.
__global__ void __launch_bounds__(1024)
k_fc6(const float* __restrict__ w, const float* __restrict__ bsrc,
      const float* __restrict__ xv, const float* __restrict__ sc,
      float* __restrict__ out) {
  __shared__ float rmn[1024], rmx[1024], rs[1024];
  __shared__ float s_wsf;
  const int row = blockIdx.x, tid = threadIdx.x;
  const float4* wp = (const float4*)(w + (size_t)row * 524288);
  const float4* xp = (const float4*)xv;

  float mn = __builtin_inff(), mx = -__builtin_inff();
  for (int i = tid; i < 131072; i += 1024) {
    if (i + 4096 < 131072) __builtin_prefetch(wp + i + 4096, 0, 1);
    float4 v = wp[i];
    mn = fminf(mn, fminf(fminf(v.x, v.y), fminf(v.z, v.w)));
    mx = fmaxf(mx, fmaxf(fmaxf(v.x, v.y), fmaxf(v.z, v.w)));
  }
  rmn[tid] = mn; rmx[tid] = mx;
  __syncthreads();
  for (int s = 512; s > 0; s >>= 1) {
    if (tid < s) {
      rmn[tid] = fminf(rmn[tid], rmn[tid + s]);
      rmx[tid] = fmaxf(rmx[tid], rmx[tid + s]);
    }
    __syncthreads();
  }
  if (tid == 0)
    s_wsf = fmaxf(fmaxf(fabsf(rmn[0]), fabsf(rmx[0])), 1e-8f) / 63.0f;
  __syncthreads();
  const float wsf = s_wsf;

  float acc = 0.0f;
  for (int i = tid; i < 131072; i += 1024) {
    float4 v = wp[i];
    float4 u = xp[i];
    acc += qround_clip(v.x / wsf) * u.x + qround_clip(v.y / wsf) * u.y +
           qround_clip(v.z / wsf) * u.z + qround_clip(v.w / wsf) * u.w;
  }
  rs[tid] = acc;
  __syncthreads();
  for (int s = 512; s > 0; s >>= 1) {
    if (tid < s) rs[tid] += rs[tid + s];
    __syncthreads();
  }
  if (tid == 0) {
    float bs = wsf * sc[0];
    out[row] = (rs[0] + qround_clip(bsrc[row] / bs)) * bs;
  }
}

// ------------------------------- fc8 ---------------------------------------
__global__ void k_fc8(const float* __restrict__ w, const float* __restrict__ bsrc,
                      const float* __restrict__ x6, const float* __restrict__ sc,
                      float* __restrict__ out) {
  __shared__ float rmn[128], rmx[128], rs[128];
  const int tid = threadIdx.x;
  const float psf = sc[0];
  const float xi = x6[tid] / psf;
  for (int r = 0; r < 5; ++r) {
    float v = w[r * 128 + tid];
    rmn[tid] = v; rmx[tid] = v;
    __syncthreads();
    for (int s = 64; s > 0; s >>= 1) {
      if (tid < s) {
        rmn[tid] = fminf(rmn[tid], rmn[tid + s]);
        rmx[tid] = fmaxf(rmx[tid], rmx[tid + s]);
      }
      __syncthreads();
    }
    float wsf = fmaxf(fmaxf(fabsf(rmn[0]), fabsf(rmx[0])), 1e-8f) / 63.0f;
    rs[tid] = qround_clip(v / wsf) * xi;
    __syncthreads();
    for (int s = 64; s > 0; s >>= 1) {
      if (tid < s) rs[tid] += rs[tid + s];
      __syncthreads();
    }
    if (tid == 0) {
      float bs = wsf * psf;
      out[r] = (rs[0] + qround_clip(bsrc[r] / bs)) * bs;
    }
    __syncthreads();
  }
}

// ----------------------------- launcher ------------------------------------

extern "C" void kernel_launch(void* const* d_in, const int* in_sizes, int n_in,
                              void* d_out, int out_size, void* d_ws, size_t ws_size,
                              hipStream_t stream) {
  (void)in_sizes; (void)n_in; (void)out_size; (void)ws_size;

  const float* x  = (const float*)d_in[0];
  const float* w0 = (const float*)d_in[1];
  const float* b0 = (const float*)d_in[2];
  const float* w2 = (const float*)d_in[3];
  const float* b2 = (const float*)d_in[4];
  const float* w6 = (const float*)d_in[5];
  const float* b6 = (const float*)d_in[6];
  const float* w8 = (const float*)d_in[7];
  const float* b8 = (const float*)d_in[8];
  float* out = (float*)d_out;

  // Workspace carving (~19 MB, 256B aligned slices).
  char* p = (char*)d_ws;
  auto carve = [&](size_t bytes) -> char* {
    char* r = p;
    p += (bytes + 255) & ~(size_t)255;
    return r;
  };
  float*    sc   = (float*)carve(64);                        // p_sf, 1/p_sf
  _Float16* xq   = (_Float16*)carve((size_t)3 * 16384 * 2);  // quantized input
  _Float16* w0q  = (_Float16*)carve((size_t)64 * 32 * 2);    // K padded to 32
  float*    b0i  = (float*)carve(64 * 4);
  float*    b0s  = (float*)carve(64 * 4);
  _Float16* w2q  = (_Float16*)carve((size_t)128 * 576 * 2);
  float*    b2i  = (float*)carve(128 * 4);
  float*    b2s  = (float*)carve(128 * 4);
  float*    h1   = (float*)carve((size_t)64 * 16384 * 4);
  _Float16* hip_ = (_Float16*)carve((size_t)64 * 16384 * 2);
  _Float16* lop_ = (_Float16*)carve((size_t)64 * 16384 * 2);
  float*    h2   = (float*)carve((size_t)128 * 16384 * 4);
  float*    xv   = (float*)carve((size_t)524288 * 4);
  float*    x6   = (float*)carve(128 * 4);

  k_minmax <<<1,    1024, 0, stream>>>(x, 3 * 16384, sc);
  k_quant_x<<<192,  256,  0, stream>>>(x, sc, xq, 3 * 16384);
  k_prep_w0<<<64,   32,   0, stream>>>(w0, b0, sc, w0q, b0i, b0s);
  k_prep_w2<<<128,  32,   0, stream>>>(w2, b2, sc, w2q, b2i, b2s);
  k_conv0  <<<1024, 128,  0, stream>>>(xq, w0q, b0i, b0s, h1);
  k_split  <<<4096, 256,  0, stream>>>(h1, sc, hip_, lop_, 64 * 16384);
  k_conv1  <<<1024, 256,  0, stream>>>(hip_, lop_, w2q, b2i, b2s, h2);
  k_pool   <<<2048, 256,  0, stream>>>(h2, sc, xv, 524288);
  k_fc6    <<<128,  1024, 0, stream>>>(w6, b6, xv, sc, x6);
  k_fc8    <<<1,    128,  0, stream>>>(w8, b8, x6, sc, out);
}
